// MultiHeadAttention_60541859004478
// MI455X (gfx1250) — compile-verified, matrix operand order checked by reading, not run on hardware
//
#include <hip/hip_runtime.h>

typedef __attribute__((ext_vector_type(16))) __bf16 v16bf;
typedef __attribute__((ext_vector_type(8)))  float  v8f;
typedef __attribute__((ext_vector_type(4)))  unsigned int v4u;
typedef __attribute__((ext_vector_type(8)))  int v8i_;
typedef __attribute__((ext_vector_type(4)))  int v4i_;
typedef unsigned int u32;

// K-index striping for 16-bit A/B WMMA fragments (16 halves per lane).
// VGPR v (=e>>1) holds: v<4 -> K = 2v + p + 8g ; v>=4 -> K = 2(v-4)+16 + p + 8g
__device__ __forceinline__ int kmap(int e, int g) {
    int v = e >> 1, p = e & 1;
    int b = (v < 4) ? (v * 2) : ((v - 4) * 2 + 16);
    return b + g * 8 + p;
}

__device__ __forceinline__ v8f wmma_bf16(v16bf a, v16bf b, v8f c) {
    return __builtin_amdgcn_wmma_f32_16x16x32_bf16(false, a, false, b,
                                                   (short)0, c, false, false);
}

// ---------------------------------------------------------------------------
// TDM: DMA a 32-row x 64-bf16 tile (row-major, 128B rows) from global memory
// into LDS with a 16B pad after each row (LDS row stride 144B = [72] __bf16).
// D# per CDNA5 ISA: group0 = {count,lds,global,type},
// group1 = {data_size=8B, pad 4 dwords every 32 dwords, dims/strides}.
// 6-arg builtin on this toolchain:
//   (uint32x4 g0, int32x8 g1, int32x4 g2, int32x4 g3, int32x8 extra, i32 cpol)
// ---------------------------------------------------------------------------
__device__ __forceinline__ void tdm_load_tile32x64(unsigned lds_off,
                                                   const __bf16* gsrc) {
    unsigned long long ga = (unsigned long long)(uintptr_t)gsrc;
    v4u g0;
    g0.x = 1u;                                           // count=1 (user D#)
    g0.y = lds_off;                                      // lds_addr (bytes)
    g0.z = (unsigned)ga;                                 // global_addr[31:0]
    g0.w = (unsigned)((ga >> 32) & 0x01ffffffu) | (2u << 30);  // [56:32], type=2
    v8i_ g1;
    // data_size=3 (8B), pad_enable, pad_interval=4 (32 dw), pad_amount=3 (4 dw)
    g1[0] = (3 << 16) | (1 << 20) | (4 << 22) | (3 << 25);
    g1[1] = (16 << 16);          // tensor_dim0 = 16 units (row = 128B)
    g1[2] = (4096 << 16);        // tensor_dim0 hi=0 | tensor_dim1 lo = 4096
    g1[3] = (16 << 16);          // tensor_dim1 hi=0 | tile_dim0 = 16 units
    g1[4] = 32;                  // tile_dim1 = 32 rows | tile_dim2 = 0
    g1[5] = 16;                  // tensor_dim0_stride = 16 units
    g1[6] = 0;                   // stride0 hi | tensor_dim1_stride lo (unused, 2D)
    g1[7] = 0;
    v4i_ z4 = {0, 0, 0, 0};
    v8i_ z8 = {0, 0, 0, 0, 0, 0, 0, 0};
    __builtin_amdgcn_tensor_load_to_lds(g0, g1, z4, z4, z8, 0);
}

// ---------------------------------------------------------------------------
// Projection GEMM: Y[m,n] = sum_k X[m,k]*W[n,k] + bias[n]
// X fp32 [8192,512], W fp32 [512,512]. Output bf16 in head-split [B,H,S,64].
// Block: 128 threads (4 waves), tile 64(M) x 64(N), K-step 32.
// ---------------------------------------------------------------------------
__global__ __launch_bounds__(128) void proj_kernel(
    const float* __restrict__ X, const float* __restrict__ W,
    const float* __restrict__ bias, __bf16* __restrict__ Yh) {
    __shared__ __attribute__((aligned(16))) __bf16 Xs[64][36];
    __shared__ __attribute__((aligned(16))) __bf16 Ws[64][36];
    const int tid = threadIdx.x, wave = tid >> 5, lane = tid & 31;
    const int lm = lane & 15, g = lane >> 4;
    const int m0 = blockIdx.x * 64, n0 = blockIdx.y * 64;

    v8f acc[4] = {};
    for (int k0 = 0; k0 < 512; k0 += 32) {
        for (int i = tid; i < 64 * 32; i += 128) {
            int r = i >> 5, c = i & 31;
            Xs[r][c] = (__bf16)X[(size_t)(m0 + r) * 512 + k0 + c];
            Ws[r][c] = (__bf16)W[(size_t)(n0 + r) * 512 + k0 + c];
            if (k0 + 32 < 512) {
                __builtin_prefetch(&X[(size_t)(m0 + r) * 512 + k0 + 32 + c], 0, 1);
                __builtin_prefetch(&W[(size_t)(n0 + r) * 512 + k0 + 32 + c], 0, 1);
            }
        }
        __syncthreads();
        v16bf a;
#pragma unroll
        for (int e = 0; e < 16; ++e) a[e] = Xs[wave * 16 + lm][kmap(e, g)];
#pragma unroll
        for (int nt = 0; nt < 4; ++nt) {
            v16bf b;
#pragma unroll
            for (int e = 0; e < 16; ++e) b[e] = Ws[nt * 16 + lm][kmap(e, g)];
            acc[nt] = wmma_bf16(a, b, acc[nt]);
        }
        __syncthreads();
    }
#pragma unroll
    for (int nt = 0; nt < 4; ++nt) {
        int n = n0 + nt * 16 + lm;
        int h = n >> 6, d = n & 63;
        float bv = bias[n];
#pragma unroll
        for (int r = 0; r < 8; ++r) {
            int m = m0 + wave * 16 + r + g * 8;
            int bb = m >> 12, s = m & 4095;
            size_t idx = (((size_t)bb * 8 + h) * 4096 + s) * 64 + d;
            Yh[idx] = (__bf16)(acc[nt][r] + bv);
        }
    }
}

// ---------------------------------------------------------------------------
// Flash attention: one block = (b,h, 64-query tile); 4 waves x 16 rows.
// Qh/Kh/Vh bf16 [B*H, S, 64]. Output bf16 concat [B, S, 512].
// K/V tiles double-buffered in LDS; Tensor Data Mover DMAs tile j+1 while
// the waves run WMMA/softmax on tile j (split TENSORcnt pipelining).
// ---------------------------------------------------------------------------
__global__ __launch_bounds__(128) void attn_kernel(
    const __bf16* __restrict__ Qh, const __bf16* __restrict__ Kh,
    const __bf16* __restrict__ Vh, __bf16* __restrict__ Cc) {
    __shared__ __attribute__((aligned(16))) __bf16 Ks[2][32][72];
    __shared__ __attribute__((aligned(16))) __bf16 Vs[2][32][72];
    __shared__ __attribute__((aligned(16))) __bf16 Ps[4][16][36];
    const int tid = threadIdx.x, wave = tid >> 5, lane = tid & 31;
    const int lm = lane & 15, g = lane >> 4;
    const int q0 = blockIdx.x * 64;
    const int bh = blockIdx.y;                 // b*8 + h
    const int bb = bh >> 3, h = bh & 7;
    const __bf16* Qb = Qh + (size_t)bh * 4096 * 64;
    const __bf16* Kb = Kh + (size_t)bh * 4096 * 64;
    const __bf16* Vb = Vh + (size_t)bh * 4096 * 64;
    const unsigned ks_off[2] = {(unsigned)(uintptr_t)&Ks[0][0][0],
                                (unsigned)(uintptr_t)&Ks[1][0][0]};
    const unsigned vs_off[2] = {(unsigned)(uintptr_t)&Vs[0][0][0],
                                (unsigned)(uintptr_t)&Vs[1][0][0]};

    // Q A-fragments for this wave's 16 rows (dk=64 -> 2 K-chunks of 32)
    v16bf aq[2];
    {
        const __bf16* qrow = Qb + (size_t)(q0 + wave * 16 + lm) * 64;
#pragma unroll
        for (int c = 0; c < 2; ++c)
#pragma unroll
            for (int e = 0; e < 16; ++e) aq[c][e] = qrow[c * 32 + kmap(e, g)];
    }

    float mrow[8], lrow[8];
#pragma unroll
    for (int r = 0; r < 8; ++r) { mrow[r] = -INFINITY; lrow[r] = 0.0f; }
    v8f o[4] = {};

    // prologue: start DMA of tile 0
    if (wave == 0) {
        tdm_load_tile32x64(ks_off[0], Kb);
        tdm_load_tile32x64(vs_off[0], Vb);
    }

    int buf = 0;
    for (int j0 = 0; j0 < 4096; j0 += 32, buf ^= 1) {
        if (wave == 0) {
            if (j0 + 32 < 4096) {
                // issue DMA for next tile into other buffer, then wait until
                // only those 2 are outstanding => current tile has landed
                tdm_load_tile32x64(ks_off[buf ^ 1], Kb + (size_t)(j0 + 32) * 64);
                tdm_load_tile32x64(vs_off[buf ^ 1], Vb + (size_t)(j0 + 32) * 64);
                __builtin_amdgcn_s_wait_tensorcnt(2);
            } else {
                __builtin_amdgcn_s_wait_tensorcnt(0);
            }
        }
        __syncthreads();

        // scores: 16 rows x 32 keys = two 16x16 C tiles
        v8f s[2] = {};
#pragma unroll
        for (int kt = 0; kt < 2; ++kt) {
#pragma unroll
            for (int c = 0; c < 2; ++c) {
                v16bf bk;
#pragma unroll
                for (int e = 0; e < 16; ++e)
                    bk[e] = Ks[buf][kt * 16 + lm][c * 32 + kmap(e, g)];
                s[kt] = wmma_bf16(aq[c], bk, s[kt]);
            }
        }

        // online softmax; P staged to LDS for C-layout -> A-layout conversion
#pragma unroll
        for (int r = 0; r < 8; ++r) {
            float t0 = s[0][r] * 0.125f;      // 1/sqrt(64)
            float t1 = s[1][r] * 0.125f;
            float mx = fmaxf(t0, t1);
#pragma unroll
            for (int off = 8; off > 0; off >>= 1)
                mx = fmaxf(mx, __shfl_xor(mx, off, 16));
            float mnew = fmaxf(mrow[r], mx);
            float p0 = __expf(t0 - mnew);
            float p1 = __expf(t1 - mnew);
            float ssum = p0 + p1;
#pragma unroll
            for (int off = 8; off > 0; off >>= 1)
                ssum += __shfl_xor(ssum, off, 16);
            float alpha = __expf(mrow[r] - mnew);
            mrow[r] = mnew;
            lrow[r] = lrow[r] * alpha + ssum;
#pragma unroll
            for (int nt = 0; nt < 4; ++nt) o[nt][r] *= alpha;
            Ps[wave][r + g * 8][lm]      = (__bf16)p0;
            Ps[wave][r + g * 8][lm + 16] = (__bf16)p1;
        }
        asm volatile("s_wait_dscnt 0" ::: "memory");  // wave-local DS RAW

        // P(16x32) @ V(32x64)
        v16bf pa;
#pragma unroll
        for (int e = 0; e < 16; ++e) pa[e] = Ps[wave][lm][kmap(e, g)];
#pragma unroll
        for (int nt = 0; nt < 4; ++nt) {
            v16bf bv;
#pragma unroll
            for (int e = 0; e < 16; ++e) bv[e] = Vs[buf][kmap(e, g)][nt * 16 + lm];
            o[nt] = wmma_bf16(pa, bv, o[nt]);
        }
        __syncthreads();   // all waves done with buf before it is re-targeted
    }

    // normalize and write concat [B,S,512]
#pragma unroll
    for (int nt = 0; nt < 4; ++nt) {
        int col = h * 64 + nt * 16 + lm;
#pragma unroll
        for (int r = 0; r < 8; ++r) {
            int q = q0 + wave * 16 + r + g * 8;
            float inv = 1.0f / lrow[r];
            Cc[((size_t)bb * 4096 + q) * 512 + col] = (__bf16)(o[nt][r] * inv);
        }
    }
}

// ---------------------------------------------------------------------------
// Output projection: out[m,n] = sum_k C[m,k]*Wo[n,k] + bo[n]  (fp32 out)
// ---------------------------------------------------------------------------
__global__ __launch_bounds__(128) void oproj_kernel(
    const __bf16* __restrict__ X, const float* __restrict__ W,
    const float* __restrict__ bias, float* __restrict__ Y) {
    __shared__ __attribute__((aligned(16))) __bf16 Xs[64][36];
    __shared__ __attribute__((aligned(16))) __bf16 Ws[64][36];
    const int tid = threadIdx.x, wave = tid >> 5, lane = tid & 31;
    const int lm = lane & 15, g = lane >> 4;
    const int m0 = blockIdx.x * 64, n0 = blockIdx.y * 64;

    v8f acc[4] = {};
    for (int k0 = 0; k0 < 512; k0 += 32) {
        for (int i = tid; i < 64 * 32; i += 128) {
            int r = i >> 5, c = i & 31;
            Xs[r][c] = X[(size_t)(m0 + r) * 512 + k0 + c];
            Ws[r][c] = (__bf16)W[(size_t)(n0 + r) * 512 + k0 + c];
            if (k0 + 32 < 512) {
                __builtin_prefetch(&X[(size_t)(m0 + r) * 512 + k0 + 32 + c], 0, 1);
                __builtin_prefetch(&W[(size_t)(n0 + r) * 512 + k0 + 32 + c], 0, 1);
            }
        }
        __syncthreads();
        v16bf a;
#pragma unroll
        for (int e = 0; e < 16; ++e) a[e] = Xs[wave * 16 + lm][kmap(e, g)];
#pragma unroll
        for (int nt = 0; nt < 4; ++nt) {
            v16bf b;
#pragma unroll
            for (int e = 0; e < 16; ++e) b[e] = Ws[nt * 16 + lm][kmap(e, g)];
            acc[nt] = wmma_bf16(a, b, acc[nt]);
        }
        __syncthreads();
    }
#pragma unroll
    for (int nt = 0; nt < 4; ++nt) {
        int n = n0 + nt * 16 + lm;
        float bv = bias[n];
#pragma unroll
        for (int r = 0; r < 8; ++r) {
            int m = m0 + wave * 16 + r + g * 8;
            Y[(size_t)m * 512 + n] = acc[nt][r] + bv;
        }
    }
}

extern "C" void kernel_launch(void* const* d_in, const int* in_sizes, int n_in,
                              void* d_out, int out_size, void* d_ws, size_t ws_size,
                              hipStream_t stream) {
    const float* q  = (const float*)d_in[0];
    const float* k  = (const float*)d_in[1];
    const float* v  = (const float*)d_in[2];
    const float* Wq = (const float*)d_in[3];
    const float* bq = (const float*)d_in[4];
    const float* Wk = (const float*)d_in[5];
    const float* bk = (const float*)d_in[6];
    const float* Wv = (const float*)d_in[7];
    const float* bv = (const float*)d_in[8];
    const float* Wo = (const float*)d_in[9];
    const float* bo = (const float*)d_in[10];
    float* out = (float*)d_out;

    const size_t elems = (size_t)2 * 4096 * 512;   // 4M bf16 each
    __bf16* Qh = (__bf16*)d_ws;
    __bf16* Kh = Qh + elems;
    __bf16* Vh = Kh + elems;
    __bf16* Cc = Vh + elems;

    dim3 blk(128);
    dim3 gproj(128, 8);            // 8192/64 x 512/64
    proj_kernel<<<gproj, blk, 0, stream>>>(q, Wq, bq, Qh);
    proj_kernel<<<gproj, blk, 0, stream>>>(k, Wk, bk, Kh);
    proj_kernel<<<gproj, blk, 0, stream>>>(v, Wv, bv, Vh);
    dim3 gattn(64, 16);            // 4096/64 q-tiles x (B*H)
    attn_kernel<<<gattn, blk, 0, stream>>>(Qh, Kh, Vh, Cc);
    dim3 go(128, 8);
    oproj_kernel<<<go, blk, 0, stream>>>(Cc, Wo, bo, out);
}
